// WaveletFusionBlock_39204461478378
// MI455X (gfx1250) — matrix operand order repaired and checked by reading, not compiled, not run on hardware
//
#include <hip/hip_runtime.h>

typedef __bf16 bf16_t;
typedef __attribute__((ext_vector_type(2)))  __bf16 bf16x2;
typedef __attribute__((ext_vector_type(16))) __bf16 v16bf;
typedef __attribute__((ext_vector_type(8)))  float  v8f;

#define HW   16384   // 128*128 pixels per image
#define IMGW 128

union FragB { v16bf v; int4 q[2]; };

__device__ __forceinline__ v16bf load_frag(const bf16_t* p) {
    FragB f;
    f.q[0] = ((const int4*)p)[0];
    f.q[1] = ((const int4*)p)[1];
    return f.v;
}

// ---------------------------------------------------------------------------
// Pack an (M x K) fp32 weight matrix (optionally with `ntaps` interleaved 3x3
// taps, src[(m*K+k)*ntaps + tap]) into bf16 WMMA A-fragment order:
//   dst[(((tap*MT + mt)*KT + kt)*32 + lane)*16 + e]
// A 16x32 bf16 layout: lane half h, element e -> k = kt*32 + h*16 + e
// ---------------------------------------------------------------------------
__global__ __launch_bounds__(256)
void pack_a_kernel(const float* __restrict__ src, bf16_t* __restrict__ dst,
                   int M, int K, int ntaps)
{
    const int KT = K >> 5, MT = M >> 4;
    const int total = ntaps * MT * KT * 32;
    for (int id = blockIdx.x * 256 + threadIdx.x; id < total; id += gridDim.x * 256) {
        const int lane = id & 31;
        int rest = id >> 5;
        const int kt = rest % KT; rest /= KT;
        const int mt = rest % MT;
        const int tap = rest / MT;
        const int m = mt * 16 + (lane & 15);
        const int half = lane >> 4;
        bf16_t* d = dst + (size_t)id * 16;
#pragma unroll
        for (int e = 0; e < 16; ++e) {
            const int k = kt * 32 + half * 16 + e;
            d[e] = (bf16_t)src[((size_t)m * K + k) * ntaps + tap];
        }
    }
}

// ---------------------------------------------------------------------------
// Generic 1x1-conv GEMM: Out[b,m,p] = bias[m] + sum_k W[m,k] * prep(k, p)
// Block: 256 threads = 8 waves. Tile: M=128 (wave -> 16 rows), N=64 pixels.
// K staged to LDS in 256-channel chunks as bf16 [n][k], stride 264 (+8 pad:
// DWORD stride 132 === 4 mod 64 -> conflict-free ds_load_b128 fragment reads).
// Staging: float4 global loads of a channel PAIR, packed bf16x2 LDS stores.
// MODE 0: prep = concat(feat_a, feat_b)            (K=512)  -> init_fused
// MODE 1: prep = (1-fw[ch])*X0 + fw[ch]*X1         (K=256)  -> fused
// MODE 2: prep = X0 + X1                           (K=256)  -> spatial
// MODE 3: prep = sw[p]*(X0 + X1); epi += rs*resid  (K=256)  -> out
// ---------------------------------------------------------------------------
template<int KDIM, int MODE>
__global__ __launch_bounds__(256)
void gemm1x1_kernel(const float* __restrict__ X0, const float* __restrict__ X1,
                    const float* __restrict__ scal, const bf16_t* __restrict__ Apack,
                    const float* __restrict__ bias, const float* __restrict__ resid,
                    const float* __restrict__ res_scale, float* __restrict__ Out)
{
    constexpr int LDB = 264;
    constexpr int NCH = KDIM / 256;
    __shared__ __align__(16) bf16_t Bs[64 * LDB];
    __shared__ float fwTile[256];
    __shared__ float swTile[64];

    const int tid    = threadIdx.x;
    const int b      = blockIdx.z;
    const int p0     = blockIdx.x * 64;
    const int mBlock = blockIdx.y * 128;
    const int wave = tid >> 5, lane = tid & 31;
    const int n15 = lane & 15, half = lane >> 4;

    if (MODE == 1) fwTile[tid] = scal[b * 256 + tid];
    if (MODE == 3) { if (tid < 64) swTile[tid] = scal[(size_t)b * HW + p0 + tid]; }

    v8f acc[4] = {};

    for (int c = 0; c < NCH; ++c) {
        __syncthreads();
        // stage 64 pixels x 256 channels: 2048 iters of (4 pixels x 2 channels)
        for (int idx = tid; idx < 2048; idx += 256) {
            const int n4  = (idx & 15) * 4;
            const int ch0 = (idx >> 4) * 2;
            const int p   = p0 + n4;
            float ca[4], cb[4];
            if (MODE == 0) {
                const int chG = c * 256 + ch0;
                const float* s0 = (chG < 256)
                    ? X0 + ((size_t)b * 256 + chG) * HW + p
                    : X1 + ((size_t)b * 256 + (chG - 256)) * HW + p;
                const float4 va = *(const float4*)s0;
                const float4 vb = *(const float4*)(s0 + HW);
                ca[0] = va.x; ca[1] = va.y; ca[2] = va.z; ca[3] = va.w;
                cb[0] = vb.x; cb[1] = vb.y; cb[2] = vb.z; cb[3] = vb.w;
            } else {
                const float4 a0 = *(const float4*)(X0 + ((size_t)b * 256 + ch0) * HW + p);
                const float4 a1 = *(const float4*)(X0 + ((size_t)b * 256 + ch0 + 1) * HW + p);
                const float4 b0 = *(const float4*)(X1 + ((size_t)b * 256 + ch0) * HW + p);
                const float4 b1 = *(const float4*)(X1 + ((size_t)b * 256 + ch0 + 1) * HW + p);
                const float a0_[4] = {a0.x, a0.y, a0.z, a0.w};
                const float a1_[4] = {a1.x, a1.y, a1.z, a1.w};
                const float b0_[4] = {b0.x, b0.y, b0.z, b0.w};
                const float b1_[4] = {b1.x, b1.y, b1.z, b1.w};
                if (MODE == 1) {
                    const float f0 = fwTile[ch0], f1 = fwTile[ch0 + 1];
#pragma unroll
                    for (int j = 0; j < 4; ++j) {
                        ca[j] = (1.0f - f0) * a0_[j] + f0 * b0_[j];
                        cb[j] = (1.0f - f1) * a1_[j] + f1 * b1_[j];
                    }
                } else if (MODE == 2) {
#pragma unroll
                    for (int j = 0; j < 4; ++j) { ca[j] = a0_[j] + b0_[j]; cb[j] = a1_[j] + b1_[j]; }
                } else {
#pragma unroll
                    for (int j = 0; j < 4; ++j) {
                        const float s = swTile[n4 + j];
                        ca[j] = s * (a0_[j] + b0_[j]);
                        cb[j] = s * (a1_[j] + b1_[j]);
                    }
                }
            }
#pragma unroll
            for (int j = 0; j < 4; ++j) {
                bf16x2 t;
                t[0] = (bf16_t)ca[j];
                t[1] = (bf16_t)cb[j];
                *(bf16x2*)&Bs[(n4 + j) * LDB + ch0] = t;   // ds_store_b32
            }
        }
        __syncthreads();

        const int mt = (mBlock >> 4) + wave;
#pragma unroll
        for (int kt = 0; kt < 8; ++kt) {
            const int ktg = c * 8 + kt;
            const v16bf a = load_frag(Apack + ((size_t)mt * (KDIM / 32) + ktg) * 512 + lane * 16);
            const int kOff = kt * 32 + half * 16;
#pragma unroll
            for (int nc = 0; nc < 4; ++nc) {
                const v16bf bb = load_frag(&Bs[(nc * 16 + n15) * LDB + kOff]);
                acc[nc] = __builtin_amdgcn_wmma_f32_16x16x32_bf16(
                    false, a, false, bb, (short)0, acc[nc], false, false);
            }
        }
    }

    const float rs = (MODE == 3) ? res_scale[0] : 0.0f;
#pragma unroll
    for (int nc = 0; nc < 4; ++nc) {
        const int p = p0 + nc * 16 + n15;
#pragma unroll
        for (int r = 0; r < 8; ++r) {
            const int m = mBlock + wave * 16 + half * 8 + r;
            float v = acc[nc][r] + bias[m];
            if (MODE == 3) v += rs * resid[((size_t)b * 256 + m) * HW + p];
            Out[((size_t)b * 256 + m) * HW + p] = v;
        }
    }
}

// ---------------------------------------------------------------------------
// 3x3 conv as implicit GEMM: sa = relu(W_sa1 *3x3 spatial + b_sa1), M=64, K=256.
// Stages rows {y-1, y, y+1} with a 1-pixel halo (66 wide, zero-padded) ONCE
// per 128-channel K-chunk; each tap's dx shift is an LDS row offset. Only 2
// barrier pairs per block instead of 9. LDS = 3*68*136 bf16 = 55.5 KB.
// ---------------------------------------------------------------------------
__global__ __launch_bounds__(256)
void conv3x3_kernel(const float* __restrict__ Xin, const bf16_t* __restrict__ Apack,
                    const float* __restrict__ bias, float* __restrict__ Out)
{
    constexpr int LDB = 136;               // 128 + 8 pad (DWORD stride 68 === 4 mod 64)
    __shared__ __align__(16) bf16_t Brow[3 * 68 * LDB];
    const int tid = threadIdx.x;
    const int b  = blockIdx.z;
    const int y  = blockIdx.x >> 1;
    const int x0 = (blockIdx.x & 1) * 64;
    const int wave = tid >> 5, lane = tid & 31;
    const int n15 = lane & 15, half = lane >> 4;
    const int mSub = (wave & 3) * 16;
    const int ncBase = (wave >> 2) * 2;

    v8f acc0 = {}, acc1 = {};
    for (int kc = 0; kc < 2; ++kc) {       // two 128-channel K chunks
        __syncthreads();
        for (int idx = tid; idx < 3 * 66 * 128; idx += 256) {
            const int np = idx % 66;       // halo pixel index: x = x0 - 1 + np
            const int t2 = idx / 66;
            const int chL = t2 & 127;
            const int r   = t2 >> 7;       // row: y - 1 + r
            const int yy = y - 1 + r;
            const int xx = x0 - 1 + np;
            float v = 0.0f;
            if ((unsigned)yy < 128u && (unsigned)xx < 128u)
                v = Xin[((size_t)b * 256 + kc * 128 + chL) * HW + yy * IMGW + xx];
            Brow[(r * 68 + np) * LDB + chL] = (bf16_t)v;
        }
        __syncthreads();

        for (int tap = 0; tap < 9; ++tap) {
            const int dyI = tap / 3, dx = tap % 3 - 1;
            const bf16_t* At = Apack + (size_t)tap * (4 * 8 * 512)
                             + (size_t)(mSub >> 4) * (8 * 512);
#pragma unroll
            for (int kt = 0; kt < 4; ++kt) {
                const int ktg = kc * 4 + kt;
                const v16bf a = load_frag(At + ktg * 512 + lane * 16);
                const int kOff = kt * 32 + half * 16;
                const int np0 = (ncBase + 0) * 16 + n15 + dx + 1;
                const v16bf b0 = load_frag(&Brow[(dyI * 68 + np0) * LDB + kOff]);
                acc0 = __builtin_amdgcn_wmma_f32_16x16x32_bf16(false, a, false, b0, (short)0, acc0, false, false);
                const v16bf b1 = load_frag(&Brow[(dyI * 68 + np0 + 16) * LDB + kOff]);
                acc1 = __builtin_amdgcn_wmma_f32_16x16x32_bf16(false, a, false, b1, (short)0, acc1, false, false);
            }
        }
    }
#pragma unroll
    for (int r = 0; r < 8; ++r) {
        const int m = mSub + half * 8 + r;
        const float bb = bias[m];
        const int pA = y * IMGW + x0 + (ncBase + 0) * 16 + n15;
        const int pB = y * IMGW + x0 + (ncBase + 1) * 16 + n15;
        Out[((size_t)b * 64 + m) * HW + pA] = fmaxf(acc0[r] + bb, 0.0f);
        Out[((size_t)b * 64 + m) * HW + pB] = fmaxf(acc1[r] + bb, 0.0f);
    }
}

// ---------------------------------------------------------------------------
// Mod-4 parity-class sums of init_fused: Sp[b*256+ch][(r%4)*4 + (c%4)].
// float4 loads: component j has c%4 == j; r%4 == (tid>>5)&3 is fixed/thread.
// ---------------------------------------------------------------------------
__global__ __launch_bounds__(256)
void parity_kernel(const float* __restrict__ X, float* __restrict__ Sp)
{
    const int bc = blockIdx.x;
    const int tid = threadIdx.x;
    const float* src = X + (size_t)bc * HW;
    float s[4] = {0.0f, 0.0f, 0.0f, 0.0f};
    for (int k = 0; k < 16; ++k) {
        const float4 v = *(const float4*)&src[tid * 4 + 1024 * k];
        s[0] += v.x; s[1] += v.y; s[2] += v.z; s[3] += v.w;
    }
    __shared__ float part[4 * 256];
#pragma unroll
    for (int j = 0; j < 4; ++j) part[j * 256 + tid] = s[j];
    __syncthreads();
    if (tid < 16) {                        // bin = pr*4 + pc
        const int pr = tid >> 2, pc = tid & 3;
        float acc = 0.0f;
        for (int q = 0; q < 2; ++q)        // threads with (t>>5)&3 == pr
            for (int l = 0; l < 32; ++l)
                acc += part[pc * 256 + (pr + 4 * q) * 32 + l];
        Sp[(size_t)bc * 16 + tid] = acc;
    }
}

// ---------------------------------------------------------------------------
// freq (analytic 2-level DWT pooling on adj = W_adj@xs + b_adj) + FC -> fw.
// One block per batch. thread t: group g = t>>4, channel o = t&15.
// ---------------------------------------------------------------------------
__global__ __launch_bounds__(256)
void freqfc_kernel(const float* __restrict__ Sp, const float* __restrict__ W_adj,
                   const float* __restrict__ b_adj, const float* __restrict__ subband,
                   const float* __restrict__ W_fa1, const float* __restrict__ b_fa1,
                   const float* __restrict__ W_fa2, const float* __restrict__ b_fa2,
                   float* __restrict__ fw)
{
    const int b = blockIdx.x, tid = threadIdx.x;
    __shared__ float SpL[256 * 16];
    __shared__ float freqL[256];
    __shared__ float h1L[64];
    for (int idx = tid; idx < 4096; idx += 256) SpL[idx] = Sp[(size_t)b * 4096 + idx];
    __syncthreads();

    const float sb0 = subband[0], sb1 = subband[1], sb2 = subband[2], sb3 = subband[3];
    const float mx = fmaxf(fmaxf(sb0, sb1), fmaxf(sb2, sb3));
    const float e0 = expf(sb0 - mx), e1 = expf(sb1 - mx), e2 = expf(sb2 - mx), e3 = expf(sb3 - mx);
    const float inv = 1.0f / (e0 + e1 + e2 + e3);
    const float w0 = e0 * inv, w1 = e1 * inv, w2 = e2 * inv, w3 = e3 * inv;

    const int g = tid >> 4, o = tid & 15;
    float spa[16];
#pragma unroll
    for (int par = 0; par < 16; ++par) {
        float s = b_adj[o] * 1024.0f;      // 1024 pixels per mod-4 class
#pragma unroll
        for (int i = 0; i < 16; ++i)
            s += W_adj[o * 16 + i] * SpL[(g * 16 + i) * 16 + par];
        spa[par] = s;
    }
#define S4(r0_, r1_, c0_, c1_) (spa[(r0_)*4+(c0_)] + spa[(r0_)*4+(c1_)] + spa[(r1_)*4+(c0_)] + spa[(r1_)*4+(c1_)])
    const float m1 = S4(0, 2, 0, 2) * (1.0f / 8192.0f);
    const float m2 = S4(1, 3, 0, 2) * (1.0f / 8192.0f);
    const float m3 = S4(0, 2, 1, 3) * (1.0f / 8192.0f);
    const float m4 = S4(1, 3, 1, 3) * (1.0f / 8192.0f);
    const float lf0 = w0 * (m1 + m2 + m3 + m4) + w1 * (-m1 - m2 + m3 + m4)
                    + w2 * (-m1 + m2 - m3 + m4) + w3 * (m1 - m2 - m3 + m4);
    const float n1 = S4(0, 1, 0, 1) * (1.0f / 4096.0f);
    const float n2 = S4(2, 3, 0, 1) * (1.0f / 4096.0f);
    const float n3 = S4(0, 1, 2, 3) * (1.0f / 4096.0f);
    const float n4 = S4(2, 3, 2, 3) * (1.0f / 4096.0f);
    const float lf1 = w0 * (n1 + n2 + n3 + n4) + w1 * (-n1 - n2 + n3 + n4)
                    + w2 * (-n1 + n2 - n3 + n4) + w3 * (n1 - n2 - n3 + n4);
#undef S4
    freqL[g * 16 + o] = lf0 * (1.0f / 3.0f) + lf1 * (2.0f / 3.0f);   // lw = [1/3, 2/3]
    __syncthreads();

    if (tid < 64) {
        float s = b_fa1[tid];
        for (int j = 0; j < 256; ++j) s += W_fa1[tid * 256 + j] * freqL[j];
        h1L[tid] = fmaxf(s, 0.0f);
    }
    __syncthreads();
    {
        float s = b_fa2[tid];
        for (int j = 0; j < 64; ++j) s += W_fa2[tid * 64 + j] * h1L[j];
        fw[(size_t)b * 256 + tid] = 1.0f / (1.0f + expf(-s));
    }
}

// sw[b,p] = sigmoid(b_sa2 + sum_k W_sa2[k] * sa[b,k,p]) ; 4 pixels per thread
__global__ __launch_bounds__(256)
void sw_kernel(const float* __restrict__ sa, const float* __restrict__ W_sa2,
               const float* __restrict__ b_sa2, float* __restrict__ sw)
{
    const int q = blockIdx.x * 256 + threadIdx.x;   // quad index, 0 .. 8*HW/4-1
    const int b = q >> 12, pp = (q & 4095) * 4;
    const float bias = b_sa2[0];
    float s0 = bias, s1 = bias, s2 = bias, s3 = bias;
#pragma unroll 8
    for (int k = 0; k < 64; ++k) {
        const float w = W_sa2[k];
        const float4 v = *(const float4*)&sa[((size_t)b * 64 + k) * HW + pp];
        s0 += w * v.x; s1 += w * v.y; s2 += w * v.z; s3 += w * v.w;
    }
    float4 r;
    r.x = 1.0f / (1.0f + expf(-s0));
    r.y = 1.0f / (1.0f + expf(-s1));
    r.z = 1.0f / (1.0f + expf(-s2));
    r.w = 1.0f / (1.0f + expf(-s3));
    *(float4*)&sw[(size_t)b * HW + pp] = r;
}

// ---------------------------------------------------------------------------
extern "C" void kernel_launch(void* const* d_in, const int* in_sizes, int n_in,
                              void* d_out, int out_size, void* d_ws, size_t ws_size,
                              hipStream_t stream)
{
    (void)in_sizes; (void)n_in; (void)out_size; (void)ws_size;
    const float* feat_a  = (const float*)d_in[0];
    const float* feat_b  = (const float*)d_in[1];
    const float* W_fuse  = (const float*)d_in[2];
    const float* b_fuse  = (const float*)d_in[3];
    const float* W_adj   = (const float*)d_in[4];
    const float* b_adj   = (const float*)d_in[5];
    const float* subband = (const float*)d_in[6];
    const float* W_fa1   = (const float*)d_in[7];
    const float* b_fa1   = (const float*)d_in[8];
    const float* W_fa2   = (const float*)d_in[9];
    const float* b_fa2   = (const float*)d_in[10];
    const float* W_ch    = (const float*)d_in[11];
    const float* b_ch    = (const float*)d_in[12];
    const float* W_sa1   = (const float*)d_in[13];
    const float* b_sa1   = (const float*)d_in[14];
    const float* W_sa2   = (const float*)d_in[15];
    const float* b_sa2   = (const float*)d_in[16];
    const float* W_out   = (const float*)d_in[17];
    const float* b_out   = (const float*)d_in[18];
    const float* res_scale = (const float*)d_in[19];

    char* ws = (char*)d_ws;
    size_t off = 0;
    auto alloc = [&](size_t bytes) { size_t r = off; off += (bytes + 255) & ~(size_t)255; return r; };
    float*  initF  = (float*)(ws + alloc((size_t)8 * 256 * HW * 4));   // 134 MB
    float*  fusedB = (float*)(ws + alloc((size_t)8 * 256 * HW * 4));   // 134 MB
    float*  saBuf  = (float*)(ws + alloc((size_t)8 * 64 * HW * 4));    // 33.5 MB
    float*  swBuf  = (float*)(ws + alloc((size_t)8 * HW * 4));
    float*  SpBuf  = (float*)(ws + alloc((size_t)8 * 256 * 16 * 4));
    float*  fwBuf  = (float*)(ws + alloc((size_t)8 * 256 * 4));
    bf16_t* paFuse = (bf16_t*)(ws + alloc((size_t)16 * 16 * 512 * 2));
    bf16_t* paCh   = (bf16_t*)(ws + alloc((size_t)16 * 8 * 512 * 2));
    bf16_t* paOut  = (bf16_t*)(ws + alloc((size_t)16 * 8 * 512 * 2));
    bf16_t* paSa1  = (bf16_t*)(ws + alloc((size_t)9 * 4 * 8 * 512 * 2));
    float*  spatial = (float*)d_out;   // lives in d_out until consumed, then overwritten

    pack_a_kernel<<<32, 256, 0, stream>>>(W_fuse, paFuse, 256, 512, 1);
    pack_a_kernel<<<16, 256, 0, stream>>>(W_ch,   paCh,   256, 256, 1);
    pack_a_kernel<<<16, 256, 0, stream>>>(W_out,  paOut,  256, 256, 1);
    pack_a_kernel<<<36, 256, 0, stream>>>(W_sa1,  paSa1,   64, 256, 9);

    const dim3 g(HW / 64, 2, 8);   // 256 pixel-tiles x 2 M-tiles x batch
    gemm1x1_kernel<512, 0><<<g, 256, 0, stream>>>(feat_a, feat_b, nullptr, paFuse, b_fuse,
                                                  nullptr, nullptr, initF);
    parity_kernel<<<8 * 256, 256, 0, stream>>>(initF, SpBuf);
    freqfc_kernel<<<8, 256, 0, stream>>>(SpBuf, W_adj, b_adj, subband,
                                         W_fa1, b_fa1, W_fa2, b_fa2, fwBuf);
    gemm1x1_kernel<256, 1><<<g, 256, 0, stream>>>(feat_a, feat_b, fwBuf, paCh, b_ch,
                                                  nullptr, nullptr, fusedB);
    gemm1x1_kernel<256, 2><<<g, 256, 0, stream>>>(fusedB, initF, nullptr, paCh, b_ch,
                                                  nullptr, nullptr, spatial);
    conv3x3_kernel<<<dim3(256, 1, 8), 256, 0, stream>>>(spatial, paSa1, b_sa1, saBuf);
    sw_kernel<<<8 * HW / 1024, 256, 0, stream>>>(saBuf, W_sa2, b_sa2, swBuf);
    gemm1x1_kernel<256, 3><<<g, 256, 0, stream>>>(fusedB, feat_b, swBuf, paOut, b_out,
                                                  initF, res_scale, (float*)d_out);
}
